// CausalSelfAttention_71227737636918
// MI455X (gfx1250) — matrix operand order repaired and checked by reading, not compiled, not run on hardware
//
#include <hip/hip_runtime.h>

// Problem constants (CausalSelfAttention: B=4, T=2048, C=1024, H=16, D=64)
#define BB 4
#define TT 2048
#define CC 1024
#define HH 16
#define DD 64

typedef _Float16 f16;
typedef __attribute__((ext_vector_type(16))) _Float16 v16h;
typedef __attribute__((ext_vector_type(8)))  _Float16 h8;
typedef __attribute__((ext_vector_type(4)))  _Float16 h4;
typedef __attribute__((ext_vector_type(8)))  float    v8f;
typedef __attribute__((ext_vector_type(4)))  float    f4;
typedef __attribute__((ext_vector_type(4)))  unsigned int v4u;
typedef __attribute__((ext_vector_type(8)))  int      v8i;
typedef __attribute__((ext_vector_type(4)))  int      v4i;

static __device__ __forceinline__ v16h cat16(h8 lo, h8 hi) {
  return __builtin_shufflevector(lo, hi, 0,1,2,3,4,5,6,7,8,9,10,11,12,13,14,15);
}
static __device__ __forceinline__ v8f wmma16(v16h a, v16h b, v8f c) {
  // D = A(16x32 f16) * B(32x16 f16) + C(16x16 f32)
  return __builtin_amdgcn_wmma_f32_16x16x32_f16(false, a, false, b, (short)0, c, false, false);
}

// ---- CDNA5 async global->LDS copy (ASYNCcnt) ------------------------------
static __device__ __forceinline__ void async_copy_b128(unsigned int lds_off, unsigned long long gaddr) {
  asm volatile("global_load_async_to_lds_b128 %0, %1, off" :: "v"(lds_off), "v"(gaddr) : "memory");
}
static __device__ __forceinline__ void wait_async0() {
  asm volatile("s_wait_asynccnt 0" ::: "memory");
}

// ---- CDNA5 Tensor Data Mover: 2D tile global->LDS (TENSORcnt) -------------
static __device__ __forceinline__ void tdm_load_2d(unsigned int lds_off, unsigned long long gaddr,
                                                   int tensor_d0, int tensor_d1,
                                                   int tile_d0, int tile_d1,
                                                   int d0_stride,
                                                   int pad_interval_enum, int pad_amount_enum) {
  v4u g0;
  g0[0] = 1u;                                       // count=1, user descriptor
  g0[1] = lds_off;                                  // lds_addr (bytes)
  g0[2] = (unsigned int)gaddr;                      // global_addr[31:0]
  g0[3] = (unsigned int)(gaddr >> 32) | (2u << 30); // global_addr[56:32] | type=2
  v8i g1;
  g1[0] = (1 << 16)                                 // data_size = 2 bytes
        | (1 << 20)                                 // pad_enable
        | (pad_interval_enum << 22)
        | (pad_amount_enum << 25);
  g1[1] = (tensor_d0 & 0xFFFF) << 16;               // tensor_dim0[15:0]
  g1[2] = ((tensor_d0 >> 16) & 0xFFFF)              // tensor_dim0[31:16]
        | ((tensor_d1 & 0xFFFF) << 16);             // tensor_dim1[15:0]
  g1[3] = ((tensor_d1 >> 16) & 0xFFFF)              // tensor_dim1[31:16]
        | ((tile_d0 & 0xFFFF) << 16);               // tile_dim0
  g1[4] = (tile_d1 & 0xFFFF);                       // tile_dim1 (tile_dim2=0)
  g1[5] = d0_stride;                                // tensor_dim0_stride[31:0]
  g1[6] = 0;
  g1[7] = 0;
  v4i z4 = (v4i)0;
#if __has_include(<hip/amd_detail/amd_gfx1250_TDM.h>)
  v8i z8 = (v8i)0;
  __builtin_amdgcn_tensor_load_to_lds(g0, g1, z4, z4, z8, 0);
#else
  __builtin_amdgcn_tensor_load_to_lds(g0, g1, z4, z4, 0);
#endif
}

// ---------------------------------------------------------------------------
// f32 -> f16 pre-conversion (weights reused 64x per GEMM; convert once)
// ---------------------------------------------------------------------------
__global__ __launch_bounds__(256)
void cvt_f32_to_f16(const float* __restrict__ src, f16* __restrict__ dst, int n4) {
  const int i = blockIdx.x * blockDim.x + threadIdx.x;
  if (i < n4) {
    f4 v = *(const f4*)(src + (size_t)i * 4);
    *(h4*)(dst + (size_t)i * 4) = __builtin_convertvector(v, h4);
  }
}

// ---------------------------------------------------------------------------
// Double-buffered tiled GEMM: C[M,N] = A[M,K]*B[K,N], f16 ops, f32 accum.
// Block 128x128, 8 waves (2x4); wave = 64x32 = 4x2 WMMA tiles; K-step 32.
// Tile k+32 is staged (A: async global->LDS DMA; B: manual transpose) while
// WMMA runs on tile k. One barrier per K-step.
//   EPI=0: f32 C;  EPI=1: scatter QKV -> Q/K [b][h][t][d], V [b][h][d][t]
// ---------------------------------------------------------------------------
template <int EPI>
__global__ __launch_bounds__(256)
void gemm_wmma(const f16* __restrict__ Ah, const f16* __restrict__ Bh,
               float* __restrict__ Cf,
               f16* __restrict__ Qh, f16* __restrict__ Kh, f16* __restrict__ Vh,
               int M, int N, int K)
{
  __shared__ __align__(16) f16 As[2][128 * 40];   // [m][k], stride 40 halves
  __shared__ __align__(16) f16 Bs[2][128 * 40];   // [n][k], stride 40 halves

  const int tid  = threadIdx.x;
  const int m0   = blockIdx.y * 128;
  const int nblk = blockIdx.x * 128;
  const int w    = tid >> 5;
  const int lane = tid & 31;
  const int lhi  = lane >> 4;
  const int llo  = lane & 15;
  const int wr   = w >> 2;          // 0..1 -> 64-row slab
  const int wc   = w & 3;           // 0..3 -> 32-col slab

  v8f acc[4][2];
#pragma unroll
  for (int i = 0; i < 4; ++i)
#pragma unroll
    for (int j = 0; j < 2; ++j)
#pragma unroll
      for (int r = 0; r < 8; ++r) acc[i][j][r] = 0.0f;

  // A tile (128x32 halves = 8KB) via 512 async 16B chunks
  auto stageA = [&](int k0, int buf) {
#pragma unroll
    for (int p = 0; p < 2; ++p) {
      const int idx  = tid + p * 256;
      const int row  = idx >> 2;          // 4 chunks (64B) per row
      const int koff = (idx & 3) * 8;     // halves
      const f16* g = Ah + (size_t)(m0 + row) * K + k0 + koff;
      async_copy_b128((unsigned int)(uintptr_t)&As[buf][row * 40 + koff],
                      (unsigned long long)(uintptr_t)g);
    }
  };
  // B tile (32x128) transposed into Bs[n][k]
  auto stageB = [&](int k0, int buf) {
    const int kk   = tid & 31;
    const int ngrp = tid >> 5;
    const f16* src = Bh + (size_t)(k0 + kk) * N + nblk + ngrp * 16;
    if (k0 + 32 < K) __builtin_prefetch(src + (size_t)32 * N, 0, 1);
    h8 v0 = *(const h8*)src;
    h8 v1 = *(const h8*)(src + 8);
#pragma unroll
    for (int i = 0; i < 8; ++i) {
      Bs[buf][(ngrp * 16 + i) * 40 + kk]     = v0[i];
      Bs[buf][(ngrp * 16 + 8 + i) * 40 + kk] = v1[i];
    }
  };

  stageA(0, 0);
  stageB(0, 0);

  int buf = 0;
  for (int k0 = 0; k0 < K; k0 += 32, buf ^= 1) {
    wait_async0();        // tile k0 (this wave's async slice) has landed
    __syncthreads();      // publish LDS; all waves done reading buf^1
    if (k0 + 32 < K) {    // stage next tile under this tile's WMMAs
      stageA(k0 + 32, buf ^ 1);
      stageB(k0 + 32, buf ^ 1);
    }

    // ---- WMMA fragments from LDS (ISA 7.12.2 layouts) ----
    v16h afr[4];
#pragma unroll
    for (int i = 0; i < 4; ++i) {
      // A lane: row = llo; e0..7 -> K chunk lhi*8, e8..15 -> 16+lhi*8
      const f16* p = &As[buf][(wr * 64 + i * 16 + llo) * 40 + lhi * 8];
      afr[i] = cat16(*(const h8*)p, *(const h8*)(p + 16));
    }
    v16h bfr[2];
#pragma unroll
    for (int j = 0; j < 2; ++j) {
      // B lane: col = llo; e0..7 -> K chunk lhi*8, e8..15 -> 16+lhi*8
      const f16* p = &Bs[buf][(wc * 32 + j * 16 + llo) * 40 + lhi * 8];
      bfr[j] = cat16(*(const h8*)p, *(const h8*)(p + 16));
    }
#pragma unroll
    for (int i = 0; i < 4; ++i)
#pragma unroll
      for (int j = 0; j < 2; ++j)
        acc[i][j] = wmma16(afr[i], bfr[j], acc[i][j]);
  }

  // ---- epilogue: C layout row = r + 8*lhi, col = llo ----
#pragma unroll
  for (int i = 0; i < 4; ++i) {
#pragma unroll
    for (int j = 0; j < 2; ++j) {
      const int gn = nblk + wc * 32 + j * 16 + llo;
#pragma unroll
      for (int r = 0; r < 8; ++r) {
        const int gm = m0 + wr * 64 + i * 16 + r + 8 * lhi;
        const float val = acc[i][j][r];
        if (EPI == 0) {
          Cf[(size_t)gm * N + gn] = val;
        } else {
          const int seg = gn >> 10;        // 0=q, 1=k, 2=v
          const int c   = gn & 1023;
          const int hd  = c >> 6;
          const int dd  = c & 63;
          const int bb  = gm >> 11;
          const int tt  = gm & 2047;
          const size_t bh = (size_t)bb * HH + hd;
          const f16 hv = (f16)val;
          if (seg == 0)      Qh[(bh * TT + tt) * DD + dd] = hv;   // [b][h][t][d]
          else if (seg == 1) Kh[(bh * TT + tt) * DD + dd] = hv;   // [b][h][t][d]
          else               Vh[(bh * DD + dd) * TT + tt] = hv;   // [b][h][d][t]
        }
      }
    }
  }
}

// ---------------------------------------------------------------------------
// Flash attention: 8 waves / 128 query rows per workgroup. K tile (32x64) and
// transposed V tile (64x32) are TDM-staged into double-buffered LDS; block
// kb+1's TDM runs under block kb's WMMAs; one barrier per key block.
// ---------------------------------------------------------------------------
__global__ __launch_bounds__(256)
void attn_wmma(const f16* __restrict__ Qh, const f16* __restrict__ Kh,
               const f16* __restrict__ Vh, f16* __restrict__ Yh)
{
  __shared__ __align__(16) f16 Ks[2][32 * 72];    // [t_loc][d], stride 72 (TDM pad)
  __shared__ __align__(16) f16 Vs[2][64 * 40];    // [d][t_loc], stride 40 (TDM pad)
  __shared__ __align__(16) f16 Ps[8][16 * 32];    // per-wave P transpose buffer

  const int tid  = threadIdx.x;
  const int w    = tid >> 5;
  const int lane = tid & 31;
  const int lhi  = lane >> 4, llo = lane & 15;
  const int blk  = blockIdx.x;
  const int qblk = blk & 15;            // T/128 = 16 query blocks
  const int bh   = blk >> 4;            // b*16 + h
  const int hd   = bh & (HH - 1);
  const int bb   = bh >> 4;
  const int qt0  = qblk * 128 + w * 16; // this wave's query tile

  const f16* Qb = Qh + (size_t)bh * TT * DD;
  const f16* Kb = Kh + (size_t)bh * TT * DD;
  const f16* Vb = Vh + (size_t)bh * DD * TT;      // transposed [d][t]

  // Q fragments: A layout, row = qt0+llo, K-dim = d (two 32-wide halves)
  v16h aQ[2];
#pragma unroll
  for (int s = 0; s < 2; ++s) {
    const f16* p = Qb + (size_t)(qt0 + llo) * DD + s * 32 + lhi * 8;
    aQ[s] = cat16(*(const h8*)p, *(const h8*)(p + 16));
  }

  float mrow[8], lrow[8], corr[8];
  v8f accO[4];
#pragma unroll
  for (int r = 0; r < 8; ++r) { mrow[r] = -3.0e38f; lrow[r] = 0.0f; }
#pragma unroll
  for (int f = 0; f < 4; ++f)
#pragma unroll
    for (int r = 0; r < 8; ++r) accO[f][r] = 0.0f;

  const int nkb_w   = (qt0 + 47) >> 5;            // this wave's causal bound
  const int nkb_max = (qblk * 128 + 159) >> 5;    // wave 7's bound

  auto stageKV = [&](int kb, int sbuf) {
    const int kt0 = kb * 32;
    // K: tensor [T x 64], tile 64x32, pad 128B rows -> LDS stride 144B
    tdm_load_2d((unsigned int)(uintptr_t)&Ks[sbuf][0],
                (unsigned long long)(uintptr_t)(Kb + (size_t)kt0 * DD),
                /*tensor_d0=*/DD, /*tensor_d1=*/TT,
                /*tile_d0=*/DD, /*tile_d1=*/32,
                /*d0_stride=*/DD,
                /*pad_interval(32 dw)=*/4, /*pad_amount(4 dw)=*/3);
    // V^T: tensor [64 x T], tile 32x64, pad 64B rows -> LDS stride 80B
    tdm_load_2d((unsigned int)(uintptr_t)&Vs[sbuf][0],
                (unsigned long long)(uintptr_t)(Vb + kt0),
                /*tensor_d0=*/TT, /*tensor_d1=*/DD,
                /*tile_d0=*/32, /*tile_d1=*/DD,
                /*d0_stride=*/TT,
                /*pad_interval(16 dw)=*/3, /*pad_amount(4 dw)=*/3);
  };

  if (w == 0) stageKV(0, 0);

  int buf = 0;
  for (int kb = 0; kb < nkb_max; ++kb, buf ^= 1) {
    if (w == 0) __builtin_amdgcn_s_wait_tensorcnt(0);   // block kb landed
    __syncthreads();   // publish; all waves past compute(kb-1) -> buf^1 free
    if (w == 0 && kb + 1 < nkb_max) stageKV(kb + 1, buf ^ 1);  // overlap

    if (kb < nkb_w) {
      const int kt0 = kb * 32;
      // ---- scores: S = Q * K^T (two 16x16 tiles over 32 keys) ----
      v8f S0, S1;
#pragma unroll
      for (int r = 0; r < 8; ++r) { S0[r] = 0.0f; S1[r] = 0.0f; }
#pragma unroll
      for (int s = 0; s < 2; ++s) {
        const f16* p0 = &Ks[buf][(llo) * 72 + s * 32 + lhi * 8];
        const f16* p1 = &Ks[buf][(16 + llo) * 72 + s * 32 + lhi * 8];
        v16h bk0 = cat16(*(const h8*)p0, *(const h8*)(p0 + 16));
        v16h bk1 = cat16(*(const h8*)p1, *(const h8*)(p1 + 16));
        S0 = wmma16(aQ[s], bk0, S0);
        S1 = wmma16(aQ[s], bk1, S1);
      }

      const bool needmask = (kt0 + 31) > qt0;
      const float scale = 0.125f;     // 1/sqrt(64)
#pragma unroll
      for (int r = 0; r < 8; ++r) {
        float s0 = S0[r] * scale, s1 = S1[r] * scale;
        if (needmask) {
          const int qt = qt0 + r + 8 * lhi;
          if (kt0 + llo > qt)      s0 = -3.0e38f;   // causal
          if (kt0 + 16 + llo > qt) s1 = -3.0e38f;
        }
        float v = fmaxf(s0, s1);
        v = fmaxf(v, __shfl_xor(v, 1, 32));
        v = fmaxf(v, __shfl_xor(v, 2, 32));
        v = fmaxf(v, __shfl_xor(v, 4, 32));
        v = fmaxf(v, __shfl_xor(v, 8, 32));
        const float mn = fmaxf(mrow[r], v);
        corr[r] = __expf(mrow[r] - mn);
        mrow[r] = mn;
        const float p0 = __expf(s0 - mn);
        const float p1 = __expf(s1 - mn);
        S0[r] = p0; S1[r] = p1;
        float rs = p0 + p1;
        rs += __shfl_xor(rs, 1, 32);
        rs += __shfl_xor(rs, 2, 32);
        rs += __shfl_xor(rs, 4, 32);
        rs += __shfl_xor(rs, 8, 32);
        lrow[r] = lrow[r] * corr[r] + rs;
      }
#pragma unroll
      for (int f = 0; f < 4; ++f)
#pragma unroll
        for (int r = 0; r < 8; ++r) accO[f][r] *= corr[r];

      // ---- transpose P (C layout) -> A fragment via per-wave LDS ----
#pragma unroll
      for (int r = 0; r < 8; ++r) {
        const int row = r + 8 * lhi;
        Ps[w][row * 32 + llo]      = (f16)S0[r];
        Ps[w][row * 32 + 16 + llo] = (f16)S1[r];
      }
      const f16* pp = &Ps[w][llo * 32 + lhi * 8];
      v16h aP = cat16(*(const h8*)pp, *(const h8*)(pp + 16));

      // ---- O += P * V ----
#pragma unroll
      for (int f = 0; f < 4; ++f) {
        const f16* vp = &Vs[buf][(f * 16 + llo) * 40 + lhi * 8];
        v16h bv = cat16(*(const h8*)vp, *(const h8*)(vp + 16));
        accO[f] = wmma16(aP, bv, accO[f]);
      }
    }
  }

  // ---- normalize, write Y[b][t][h*64+d] as f16 for the proj GEMM ----
#pragma unroll
  for (int r = 0; r < 8; ++r) {
    const float inv = 1.0f / lrow[r];
    const int tt = qt0 + r + 8 * lhi;
    f16* yp = Yh + ((size_t)bb * TT + tt) * CC + hd * DD;
#pragma unroll
    for (int f = 0; f < 4; ++f)
      yp[f * 16 + llo] = (f16)(accO[f][r] * inv);
  }
}

// ---------------------------------------------------------------------------
extern "C" void kernel_launch(void* const* d_in, const int* in_sizes, int n_in,
                              void* d_out, int out_size, void* d_ws, size_t ws_size,
                              hipStream_t stream)
{
  (void)in_sizes; (void)n_in; (void)out_size; (void)ws_size;
  const float* x      = (const float*)d_in[0];   // [4,2048,1024]
  const float* w_attn = (const float*)d_in[1];   // [1024,3072]
  const float* w_proj = (const float*)d_in[2];   // [1024,1024]
  float* out = (float*)d_out;                    // [4,2048,1024]

  const size_t NX  = (size_t)BB * TT * CC;       // 8,388,608
  const size_t NWA = (size_t)CC * 3 * CC;        // 3,145,728
  const size_t NWP = (size_t)CC * CC;            // 1,048,576
  const size_t SEG = (size_t)BB * HH * TT * DD;  // 8,388,608

  f16* xh  = (f16*)d_ws;
  f16* wah = xh  + NX;
  f16* wph = wah + NWA;
  f16* Qh  = wph + NWP;                          // [b][h][t][d]
  f16* Kh  = Qh  + SEG;                          // [b][h][t][d]
  f16* Vh  = Kh  + SEG;                          // [b][h][d][t]
  f16* Yh  = Vh  + SEG;                          // [8192][1024]

  // 0) one-shot f32 -> f16 conversions
  cvt_f32_to_f16<<<dim3(NX  / 4 / 256), 256, 0, stream>>>(x,      xh,  NX  / 4);
  cvt_f32_to_f16<<<dim3(NWA / 4 / 256), 256, 0, stream>>>(w_attn, wah, NWA / 4);
  cvt_f32_to_f16<<<dim3(NWP / 4 / 256), 256, 0, stream>>>(w_proj, wph, NWP / 4);

  // 1) QKV = x @ w_attn, scatter into Q/K/V f16 layouts
  gemm_wmma<1><<<dim3(3 * CC / 128, BB * TT / 128), 256, 0, stream>>>(
      xh, wah, nullptr, Qh, Kh, Vh, BB * TT, 3 * CC, CC);

  // 2) flash attention: 1024 blocks x 8 waves, double-buffered TDM K/V
  attn_wmma<<<dim3(BB * HH * (TT / 128)), 256, 0, stream>>>(Qh, Kh, Vh, Yh);

  // 3) out = Y @ w_proj (f32 output)
  gemm_wmma<0><<<dim3(CC / 128, BB * TT / 128), 256, 0, stream>>>(
      Yh, wph, out, nullptr, nullptr, nullptr, BB * TT, CC, CC);
}
